// MultiHeadAttention_53790170415812
// MI455X (gfx1250) — matrix operand order repaired and checked by reading, not compiled, not run on hardware
//
#include <hip/hip_runtime.h>

typedef __bf16 bf16;
typedef __attribute__((ext_vector_type(16))) __bf16 bf16x16;
typedef __attribute__((ext_vector_type(8)))  __bf16 bf16x8;
typedef __attribute__((ext_vector_type(8)))  float  f32x8;

#define BB 2
#define SS 2048
#define EE 1024
#define HH 16
#define DD 64
// M = BB*SS = 4096, all GEMMs are 4096 x 1024 x 1024

// ---------------- WMMA fragment loaders (contiguous k = 16*half + e, same order in A and B) ----------------

// A-matrix 16x32 bf16: lane = {m = lane&15, half = lane>>4}
// element e -> k = e + 16*half
// => two contiguous 8-element (16B) loads at k0 = 16*half and k0+8.
__device__ __forceinline__ bf16x16 frag_a(const bf16* base, int lda) {
    const int lane = threadIdx.x & 31;
    const int m = lane & 15, lg = lane >> 4;
    const bf16* row = base + (size_t)m * lda + lg * 16;
    bf16x8 lo = *(const bf16x8*)(row);
    bf16x8 hi = *(const bf16x8*)(row + 8);
    bf16x16 a;
#pragma unroll
    for (int e = 0; e < 8; ++e) { a[e] = lo[e]; a[e + 8] = hi[e]; }
    return a;
}

// B-matrix 32x16 bf16 from an N-major buffer bt[n][k] (stride ldb elems):
// lane = {n = lane&15, half = lane>>4}, element e -> k = e + 16*half
__device__ __forceinline__ bf16x16 frag_b_nmajor(const bf16* base, int ldb) {
    const int lane = threadIdx.x & 31;
    const int n = lane & 15, lg = lane >> 4;
    const bf16* row = base + (size_t)n * ldb + lg * 16;
    bf16x8 lo = *(const bf16x8*)(row);
    bf16x8 hi = *(const bf16x8*)(row + 8);
    bf16x16 b;
#pragma unroll
    for (int e = 0; e < 8; ++e) { b[e] = lo[e]; b[e + 8] = hi[e]; }
    return b;
}

// B-matrix 32x16 bf16 from a K-major buffer b[k][n] (stride ldb elems).
__device__ __forceinline__ bf16x16 frag_b_kmajor(const bf16* base, int ldb) {
    const int lane = threadIdx.x & 31;
    const int n = lane & 15, lg = lane >> 4;
    bf16x16 b;
#pragma unroll
    for (int e = 0; e < 16; ++e) b[e] = base[(size_t)(e + 16 * lg) * ldb + n];
    return b;
}

#define WMMA_BF16(A, Bm, C) \
    __builtin_amdgcn_wmma_f32_16x16x32_bf16(false, (A), false, (Bm), (short)0, (C), false, false)

// CDNA5 async copy global->LDS (ASYNCcnt-tracked, bypasses VGPR round trip).
// Generic pointer to LDS: low 32 bits are the LDS byte address (ISA 10.2).
__device__ __forceinline__ void async_ld_b128(void* lds_ptr, const void* gptr) {
    unsigned loff = (unsigned)(unsigned long long)lds_ptr;
    unsigned long long ga = (unsigned long long)gptr;
    asm volatile("global_load_async_to_lds_b128 %0, %1, off"
                 :: "v"(loff), "v"(ga) : "memory");
}

// ---------------- conversion kernels ----------------

__global__ __launch_bounds__(256) void cvt_f32_bf16(const float* __restrict__ in,
                                                    bf16* __restrict__ out, int count) {
    int i = (blockIdx.x * 256 + threadIdx.x) * 4;
    if (i >= count) return;
    float4 f = *(const float4*)(in + i);
    out[i + 0] = (bf16)f.x;
    out[i + 1] = (bf16)f.y;
    out[i + 2] = (bf16)f.z;
    out[i + 3] = (bf16)f.w;
}

// W (K x N, f32 row-major) -> Wt (N x K, bf16 row-major)
__global__ __launch_bounds__(256) void transpose_cvt(const float* __restrict__ W,
                                                     bf16* __restrict__ Wt, int K, int N) {
    __shared__ float tile[32][33];
    const int k0 = blockIdx.y * 32, n0 = blockIdx.x * 32;
    const int tx = threadIdx.x & 31, ty = threadIdx.x >> 5; // 32 x 8
#pragma unroll
    for (int i = 0; i < 32; i += 8)
        tile[ty + i][tx] = W[(size_t)(k0 + ty + i) * N + n0 + tx];
    __syncthreads();
#pragma unroll
    for (int i = 0; i < 32; i += 8)
        Wt[(size_t)(n0 + ty + i) * K + k0 + tx] = (bf16)tile[tx][ty + i];
}

// ---------------- double-buffered async WMMA GEMM ----------------
// C = A(bf16 MxK) @ Bt(bf16, N-major NxK). 128x64 macro tile, 8 waves in a
// 4(M) x 2(N) grid, each wave a 32x32 tile (4 accumulators, 4 WMMA / chunk).
// Staging via global_load_async_to_lds_b128 with 2-deep LDS double buffering.

template <bool TO_BF16>
__global__ __launch_bounds__(256) void gemm_bf16_wmma(const bf16* __restrict__ A,
                                                      const bf16* __restrict__ Bt,
                                                      bf16* __restrict__ Cbf,
                                                      float* __restrict__ Cf,
                                                      const float* __restrict__ bias,
                                                      int M, int N, int K) {
    __shared__ bf16 As[2][128 * 40];
    __shared__ bf16 Bs[2][64 * 40];
    const int m0 = blockIdx.y * 128, n0 = blockIdx.x * 64;
    const int t = threadIdx.x;
    const int lr = t >> 2;          // 0..63
    const int lc = (t & 3) * 8;     // 0,8,16,24
    const int wid = t >> 5, lane = t & 31;
    const int wm = wid & 3, wn = wid >> 2;      // 4 x 2 wave grid
    const int n_ = lane & 15, lg = lane >> 4;

    // each thread moves 3 x 16B per chunk (2 A rows + 1 B row)
    auto issue = [&](int kt, int buf) {
        async_ld_b128(&As[buf][lr * 40 + lc],        A + (size_t)(m0 + lr) * K + kt + lc);
        async_ld_b128(&As[buf][(64 + lr) * 40 + lc], A + (size_t)(m0 + 64 + lr) * K + kt + lc);
        async_ld_b128(&Bs[buf][lr * 40 + lc],        Bt + (size_t)(n0 + lr) * K + kt + lc);
    };

    f32x8 acc00 = {}, acc01 = {}, acc10 = {}, acc11 = {};
    const int niter = K / 32;
    issue(0, 0);
    for (int i = 0; i < niter; ++i) {
        const int buf = i & 1;
        if (i + 1 < niter) {
            issue((i + 1) * 32, buf ^ 1);
            asm volatile("s_wait_asynccnt 3" ::: "memory");  // chunk i landed (in-order)
        } else {
            asm volatile("s_wait_asynccnt 0" ::: "memory");
        }
        __syncthreads();   // all waves' chunk-i data visible
        bf16x16 a0 = frag_a(&As[buf][(wm * 32) * 40], 40);
        bf16x16 a1 = frag_a(&As[buf][(wm * 32 + 16) * 40], 40);
        bf16x16 b0 = frag_b_nmajor(&Bs[buf][(wn * 32) * 40], 40);
        bf16x16 b1 = frag_b_nmajor(&Bs[buf][(wn * 32 + 16) * 40], 40);
        acc00 = WMMA_BF16(a0, b0, acc00);
        acc01 = WMMA_BF16(a0, b1, acc01);
        acc10 = WMMA_BF16(a1, b0, acc10);
        acc11 = WMMA_BF16(a1, b1, acc11);
        __syncthreads();   // WAR: done reading buf before it is re-issued
    }

    const int row0 = m0 + wm * 32 + 8 * lg;   // + r  (+16 for the a1 tiles)
    const int col0 = n0 + wn * 32 + n_;       // (+16 for the b1 tiles)
    if (TO_BF16) {
#pragma unroll
        for (int r = 0; r < 8; ++r) {
            Cbf[(size_t)(row0 + r) * N + col0]           = (bf16)acc00[r];
            Cbf[(size_t)(row0 + r) * N + col0 + 16]      = (bf16)acc01[r];
            Cbf[(size_t)(row0 + 16 + r) * N + col0]      = (bf16)acc10[r];
            Cbf[(size_t)(row0 + 16 + r) * N + col0 + 16] = (bf16)acc11[r];
        }
    } else {
        const float bv0 = bias[col0], bv1 = bias[col0 + 16];
#pragma unroll
        for (int r = 0; r < 8; ++r) {
            Cf[(size_t)(row0 + r) * N + col0]           = acc00[r] + bv0;
            Cf[(size_t)(row0 + r) * N + col0 + 16]      = acc01[r] + bv1;
            Cf[(size_t)(row0 + 16 + r) * N + col0]      = acc10[r] + bv0;
            Cf[(size_t)(row0 + 16 + r) * N + col0 + 16] = acc11[r] + bv1;
        }
    }
}

// ---------------- flash attention: one wave per (b, h, 16-query tile) ----------------

__global__ __launch_bounds__(32) void attention_wmma(const bf16* __restrict__ Q,
                                                     const bf16* __restrict__ Kb,
                                                     const bf16* __restrict__ Vb,
                                                     bf16* __restrict__ O) {
    const int qt = blockIdx.x;   // 0..S/16-1
    const int h  = blockIdx.y;
    const int b  = blockIdx.z;
    const int lane = threadIdx.x & 31;
    const int n_ = lane & 15, lg = lane >> 4;

    __shared__ bf16 Pl[16 * 40];   // P tile in A layout
    __shared__ bf16 Vs[32 * 72];   // V chunk [key][d]

    const size_t qbase = ((size_t)b * SS + qt * 16) * EE + h * DD;
    const bf16x16 qa0 = frag_a(Q + qbase, EE);        // d 0..31
    const bf16x16 qa1 = frag_a(Q + qbase + 32, EE);   // d 32..63

    float m_r[8], l_r[8];
    f32x8 o[4];
#pragma unroll
    for (int r = 0; r < 8; ++r) { m_r[r] = -3.0e38f; l_r[r] = 0.0f; }
#pragma unroll
    for (int tt = 0; tt < 4; ++tt) o[tt] = (f32x8){};

    const int row_base = qt * 16 + 8 * lg;  // + r
    const int jmax = (qt * 16 + 15) / 32;

    for (int j = 0; j <= jmax; ++j) {
        const int kc = j * 32;
        __syncthreads();  // WAR on Pl/Vs vs previous iteration

        // S = Q K^T : K rows are d-contiguous -> N-major B fragments
        const size_t kbase = ((size_t)b * SS + kc) * EE + h * DD;
        bf16x16 kb00 = frag_b_nmajor(Kb + kbase, EE);
        bf16x16 kb01 = frag_b_nmajor(Kb + kbase + 32, EE);
        bf16x16 kb10 = frag_b_nmajor(Kb + kbase + (size_t)16 * EE, EE);
        bf16x16 kb11 = frag_b_nmajor(Kb + kbase + (size_t)16 * EE + 32, EE);
        f32x8 s0 = {}, s1 = {};
        s0 = WMMA_BF16(qa0, kb00, s0);
        s0 = WMMA_BF16(qa1, kb01, s0);
        s1 = WMMA_BF16(qa0, kb10, s1);
        s1 = WMMA_BF16(qa1, kb11, s1);

        // stage V chunk: lane <-> key row, contiguous 16B loads
        {
            const bf16* vrow = Vb + ((size_t)b * SS + kc + lane) * EE + h * DD;
#pragma unroll
            for (int d = 0; d < DD; d += 8)
                *(bf16x8*)(Vs + lane * 72 + d) = *(const bf16x8*)(vrow + d);
        }

        // online softmax (row reductions across the 16-lane half-groups)
        float p0[8], p1[8], alpha[8];
#pragma unroll
        for (int r = 0; r < 8; ++r) {
            float a0 = s0[r] * 0.125f;       // 1/sqrt(64)
            float a1 = s1[r] * 0.125f;
            const int row = row_base + r;
            if (kc + n_ > row)      a0 = -3.0e38f;
            if (kc + 16 + n_ > row) a1 = -3.0e38f;
            float mn = fmaxf(m_r[r], fmaxf(a0, a1));
#pragma unroll
            for (int off = 1; off < 16; off <<= 1) mn = fmaxf(mn, __shfl_xor(mn, off, 32));
            const float e0 = __expf(a0 - mn);
            const float e1 = __expf(a1 - mn);
            const float al = __expf(m_r[r] - mn);
            float rs = e0 + e1;
#pragma unroll
            for (int off = 1; off < 16; off <<= 1) rs += __shfl_xor(rs, off, 32);
            l_r[r] = l_r[r] * al + rs;
            m_r[r] = mn;
            p0[r] = e0; p1[r] = e1; alpha[r] = al;
        }
#pragma unroll
        for (int tt = 0; tt < 4; ++tt)
#pragma unroll
            for (int r = 0; r < 8; ++r) o[tt][r] *= alpha[r];

        // P (C-layout) -> LDS in A layout: row m = r+8*lg, k = col-in-chunk
#pragma unroll
        for (int r = 0; r < 8; ++r) {
            Pl[(r + 8 * lg) * 40 + n_]      = (bf16)p0[r];
            Pl[(r + 8 * lg) * 40 + 16 + n_] = (bf16)p1[r];
        }
        __syncthreads();  // RAW on Pl / Vs

        // O += P @ V
        const bf16x16 pa = frag_a(Pl, 40);
#pragma unroll
        for (int tt = 0; tt < 4; ++tt) {
            bf16x16 vbf = frag_b_kmajor(Vs + tt * 16, 72);
            o[tt] = WMMA_BF16(pa, vbf, o[tt]);
        }
    }

    // normalize + write bf16 attention output at [b*S + q][h*64 + d]
    const size_t obase = ((size_t)b * SS + qt * 16 + 8 * lg) * EE + h * DD + n_;
#pragma unroll
    for (int r = 0; r < 8; ++r) {
        const float inv = 1.0f / l_r[r];
#pragma unroll
        for (int tt = 0; tt < 4; ++tt)
            O[obase + (size_t)r * EE + tt * 16] = (bf16)(o[tt][r] * inv);
    }
}

// ---------------- host launch ----------------

extern "C" void kernel_launch(void* const* d_in, const int* in_sizes, int n_in,
                              void* d_out, int out_size, void* d_ws, size_t ws_size,
                              hipStream_t stream) {
    const float* xq = (const float*)d_in[0];
    const float* xk = (const float*)d_in[1];
    const float* xv = (const float*)d_in[2];
    const float* Wq = (const float*)d_in[3];
    const float* Wk = (const float*)d_in[4];
    const float* Wv = (const float*)d_in[5];
    const float* Wp = (const float*)d_in[6];
    const float* bp = (const float*)d_in[7];
    // d_in[8] = causal mask (bool) -- implemented analytically in-kernel

    const int M = BB * SS;           // 4096
    const int N = EE;                // 1024
    const int K = EE;                // 1024
    const size_t XCNT = (size_t)M * EE;   // 4M elems
    const size_t WCNT = (size_t)EE * EE;  // 1M elems

    char* ws = (char*)d_ws;
    size_t off = 0;
    auto alloc = [&](size_t bytes) -> void* {
        void* p = ws + off;
        off += (bytes + 255) & ~(size_t)255;
        return p;
    };
    bf16* xq_b = (bf16*)alloc(XCNT * 2);
    bf16* xk_b = (bf16*)alloc(XCNT * 2);
    bf16* xv_b = (bf16*)alloc(XCNT * 2);
    bf16* wq_t = (bf16*)alloc(WCNT * 2);
    bf16* wk_t = (bf16*)alloc(WCNT * 2);
    bf16* wv_t = (bf16*)alloc(WCNT * 2);
    bf16* wp_t = (bf16*)alloc(WCNT * 2);
    bf16* Qb   = (bf16*)alloc(XCNT * 2);
    bf16* Kb   = (bf16*)alloc(XCNT * 2);
    bf16* Vb   = (bf16*)alloc(XCNT * 2);
    bf16* Ab   = (bf16*)alloc(XCNT * 2);  // attention output (bf16)
    (void)ws_size;

    // 1) convert activations
    {
        dim3 g((unsigned)(XCNT / 4 / 256));
        cvt_f32_bf16<<<g, 256, 0, stream>>>(xq, xq_b, (int)XCNT);
        cvt_f32_bf16<<<g, 256, 0, stream>>>(xk, xk_b, (int)XCNT);
        cvt_f32_bf16<<<g, 256, 0, stream>>>(xv, xv_b, (int)XCNT);
    }
    // 2) convert + transpose weights to N-major bf16
    {
        dim3 g(EE / 32, EE / 32);
        transpose_cvt<<<g, 256, 0, stream>>>(Wq, wq_t, K, N);
        transpose_cvt<<<g, 256, 0, stream>>>(Wk, wk_t, K, N);
        transpose_cvt<<<g, 256, 0, stream>>>(Wv, wv_t, K, N);
        transpose_cvt<<<g, 256, 0, stream>>>(Wp, wp_t, K, N);
    }
    // 3) QKV projections (bf16 out)
    {
        dim3 g(N / 64, M / 128);
        gemm_bf16_wmma<true><<<g, 256, 0, stream>>>(xq_b, wq_t, Qb, nullptr, nullptr, M, N, K);
        gemm_bf16_wmma<true><<<g, 256, 0, stream>>>(xk_b, wk_t, Kb, nullptr, nullptr, M, N, K);
        gemm_bf16_wmma<true><<<g, 256, 0, stream>>>(xv_b, wv_t, Vb, nullptr, nullptr, M, N, K);
    }
    // 4) causal flash attention
    {
        dim3 g(SS / 16, HH, BB);
        attention_wmma<<<g, 32, 0, stream>>>(Qb, Kb, Vb, Ab);
    }
    // 5) output projection (f32 + bias) -> d_out
    {
        dim3 g(N / 64, M / 128);
        gemm_bf16_wmma<false><<<g, 256, 0, stream>>>(Ab, wp_t, nullptr, (float*)d_out, bp, M, N, K);
    }
}
